// GanCraftGenerator_29446295781950
// MI455X (gfx1250) — matrix operand order repaired and checked by reading, not compiled, not run on hardware
//
#include <hip/hip_runtime.h>
#include <hip/hip_bf16.h>

typedef __bf16 bf16;
typedef __attribute__((ext_vector_type(16))) __bf16 v16bf;
typedef __attribute__((ext_vector_type(8)))  float  v8f;

__device__ __forceinline__ float lrelu_f(float v) { return v > 0.f ? v : 0.2f * v; }

__device__ __forceinline__ v8f wmma_bf16(v16bf a, v16bf b, v8f c) {
  // D = A(16x32 bf16) * B(32x16 bf16) + C(16x16 f32)
  return __builtin_amdgcn_wmma_f32_16x16x32_bf16(false, a, false, b, (short)0, c, false, false);
}

// ---------------------------------------------------------------------------
// Stage weights from global (row-major [OUT][K*kstride]) into LDS in the exact
// B-fragment layout: chunk c = ((ct*nk8)+k8)*32+lane holds 16 contiguous bf16.
// B element e of lane l: n = ct*16 + (l&15), k = k8*32 + (l>>4)*8 + e + (e>>3)*8.
// value = wsrc[n*orow_stride + (k)*kstride + koff]
// ---------------------------------------------------------------------------
__device__ __forceinline__ void stage_weights(bf16* wlds, const float* wsrc,
                                              int nct, int nk8, int orow_stride,
                                              int kstride, int koff,
                                              int tid, int nthreads) {
  int total = nct * nk8 * 32;
  for (int c = tid; c < total; c += nthreads) {
    int lane = c & 31;
    int k8 = (c >> 5) % nk8;
    int ct = (c >> 5) / nk8;
    int n = ct * 16 + (lane & 15);
    int kb = k8 * 32 + ((lane >> 4) << 3);
    const float* src = wsrc + n * orow_stride + koff;
    bf16* dst = wlds + c * 16;
#pragma unroll
    for (int e = 0; e < 16; ++e) dst[e] = (bf16)src[(kb + e + ((e >> 3) << 3)) * kstride];
  }
}

// A-fragment from LDS row-major bf16 activations [row][fstride], K-chunk k8.
// ISA A 16x32 layout: lanes 0-15 row=l, K={0..7,16..23}; lanes 16-31 K={8..15,24..31}.
__device__ __forceinline__ v16bf load_afrag(const bf16* fb, int fstride, int row,
                                            int k8, const float* alpha) {
  v16bf a;
  int lane = threadIdx.x & 31;
  int hoff = (lane & 16) ? 8 : 0;
  const bf16* rp = fb + row * fstride;
#pragma unroll
  for (int v = 0; v < 8; ++v) {
    int kb = k8 * 32 + ((v >> 2) << 4) + hoff + ((v & 3) << 1);
    float e0 = (float)rp[kb];
    float e1 = (float)rp[kb + 1];
    if (alpha) { e0 *= alpha[kb]; e1 *= alpha[kb + 1]; }
    a[2 * v] = (bf16)e0;
    a[2 * v + 1] = (bf16)e1;
  }
  return a;
}

// ---------------------------------------------------------------------------
// Style precompute: alpha[5][256], beta[5][256], zc[1024] -> sb
// ---------------------------------------------------------------------------
__global__ void k_style(const float* __restrict__ z,
                        const float* __restrict__ wa, const float* __restrict__ ba,
                        const float* __restrict__ wb, const float* __restrict__ bb,
                        const float* __restrict__ wz, const float* __restrict__ bz,
                        float* __restrict__ sb) {
  int t = blockIdx.x * blockDim.x + threadIdx.x;
  if (t >= 3584) return;
  const float* wr;
  float s;
  if (t < 1280) { wr = wa + t * 256; s = ba[t]; }
  else if (t < 2560) { int u = t - 1280; wr = wb + u * 256; s = bb[u]; }
  else { int u = t - 2560; wr = wz + u * 256; s = bz[u]; }
  for (int k = 0; k < 256; ++k) s += z[k] * wr[k];
  sb[t] = s;
}

// ---------------------------------------------------------------------------
// ModLinear layer (256->256): f = lrelu((f*alpha) @ W^T + beta), LDS-resident
// ---------------------------------------------------------------------------
__device__ __forceinline__ void mlp_layer(bf16* fbuf, const bf16* wlds,
                                          const float* alpha, const float* beta,
                                          int tid) {
  int wave = tid >> 5, lane = tid & 31, rtb = wave * 16;
  v16bf afr[8];
#pragma unroll
  for (int k8 = 0; k8 < 8; ++k8)
    afr[k8] = load_afrag(fbuf, 256, rtb + (lane & 15), k8, alpha);
  for (int ct = 0; ct < 16; ++ct) {
    v8f acc = {};
#pragma unroll
    for (int k8 = 0; k8 < 8; ++k8) {
      v16bf b = *((const v16bf*)wlds + (ct * 8 + k8) * 32 + lane);
      acc = wmma_bf16(afr[k8], b, acc);
    }
    int col = ct * 16 + (lane & 15);
    float be = beta[col];
#pragma unroll
    for (int v = 0; v < 8; ++v) {
      int r = rtb + v + ((lane >> 4) << 3);   // C/D layout: M = v (+8 for hi lanes)
      fbuf[r * 256 + col] = (bf16)lrelu_f(acc[v] + be);
    }
  }
}

// ---------------------------------------------------------------------------
// Fused RenderMLP + volumetric compositing.
// Block = 128 rows = 8 pixels x 16 samples, 256 threads (8 waves).
// ---------------------------------------------------------------------------
__global__ void __launch_bounds__(256) k_mlp(
    const float* __restrict__ x, const float* __restrict__ m,
    const float* __restrict__ dists,
    const float* __restrict__ w1, const float* __restrict__ b1,
    const float* __restrict__ wma, const float* __restrict__ ml_w,
    const float* __restrict__ sb,
    const float* __restrict__ wsig, const float* __restrict__ bsig,
    const float* __restrict__ wc, const float* __restrict__ bc,
    float* __restrict__ feat) {
  extern __shared__ char smem[];
  bf16* fbuf = (bf16*)smem;                          // 128x256 bf16 = 64KB
  bf16* wlds = (bf16*)(smem + 65536);                // 128KB swizzled weights
  float* cbuf = (float*)(smem + 65536 + 131072);     // 128x64 f32 = 32KB
  float* mbuf = (float*)(smem + 229376);             // 128x8
  float* sigb = (float*)(smem + 233472);             // 128
  float* dbuf = (float*)(smem + 233984);             // 128
  float* wgtb = (float*)(smem + 234496);             // 128

  int tid = threadIdx.x;
  int rowbase = blockIdx.x * 128;

  // stage x tile (bf16) + m + dists
  for (int idx = tid; idx < 128 * 128; idx += 256) {
    int r = idx >> 7, k = idx & 127;
    fbuf[r * 256 + k] = (bf16)x[(rowbase + r) * 128 + k];
  }
  if (tid < 128) {
#pragma unroll
    for (int j = 0; j < 7; ++j) mbuf[tid * 8 + j] = m[(rowbase + tid) * 7 + j];
    mbuf[tid * 8 + 7] = 0.f;
    dbuf[tid] = dists[rowbase + tid];
  }
  __syncthreads();

  // ---- fc_1: f = x@w1^T + b1 + m@wma^T, lrelu ----
  stage_weights(wlds, w1, 16, 4, 128, 1, 0, tid, 256);
  __syncthreads();
  {
    int wave = tid >> 5, lane = tid & 31, rtb = wave * 16;
    v16bf afr[4];
#pragma unroll
    for (int k8 = 0; k8 < 4; ++k8)
      afr[k8] = load_afrag(fbuf, 256, rtb + (lane & 15), k8, nullptr);
    for (int ct = 0; ct < 16; ++ct) {
      v8f acc = {};
#pragma unroll
      for (int k8 = 0; k8 < 4; ++k8) {
        v16bf b = *((const v16bf*)wlds + (ct * 4 + k8) * 32 + lane);
        acc = wmma_bf16(afr[k8], b, acc);
      }
      int col = ct * 16 + (lane & 15);
      float bb = b1[col];
      float wv[7];
#pragma unroll
      for (int j = 0; j < 7; ++j) wv[j] = wma[col * 7 + j];
#pragma unroll
      for (int v = 0; v < 8; ++v) {
        int r = rtb + v + ((lane >> 4) << 3);
        float md = bb;
#pragma unroll
        for (int j = 0; j < 7; ++j) md += mbuf[r * 8 + j] * wv[j];
        fbuf[r * 256 + col] = (bf16)lrelu_f(acc[v] + md);
      }
    }
  }

  // ---- fc_2..fc_4 (ModLinear 0..2) ----
  for (int i = 0; i < 3; ++i) {
    __syncthreads();
    stage_weights(wlds, ml_w + i * 65536, 16, 8, 256, 1, 0, tid, 256);
    __syncthreads();
    mlp_layer(fbuf, wlds, sb + i * 256, sb + 1280 + i * 256, tid);
  }

  // ---- sigma head (VALU) ----
  {
    int wave = tid >> 5, lane = tid & 31;
    int row = wave * 16 + (lane & 15);
    int k0 = (lane >> 4) * 128;
    float s = 0.f;
    for (int k = 0; k < 128; ++k) s += (float)fbuf[row * 256 + k0 + k] * wsig[k0 + k];
    s += __shfl_xor(s, 16);
    if (lane < 16) sigb[row] = s + bsig[0];
  }

  // ---- fc_5, fc_6 (ModLinear 3,4) ----
  for (int i = 3; i < 5; ++i) {
    __syncthreads();
    stage_weights(wlds, ml_w + i * 65536, 16, 8, 256, 1, 0, tid, 256);
    __syncthreads();
    mlp_layer(fbuf, wlds, sb + i * 256, sb + 1280 + i * 256, tid);
  }

  // ---- color head: c = f@wc^T + bc (64 cols) ----
  __syncthreads();
  stage_weights(wlds, wc, 4, 8, 256, 1, 0, tid, 256);
  __syncthreads();
  {
    int wave = tid >> 5, lane = tid & 31, rtb = wave * 16;
    v16bf afr[8];
#pragma unroll
    for (int k8 = 0; k8 < 8; ++k8)
      afr[k8] = load_afrag(fbuf, 256, rtb + (lane & 15), k8, nullptr);
    for (int ct = 0; ct < 4; ++ct) {
      v8f acc = {};
#pragma unroll
      for (int k8 = 0; k8 < 8; ++k8) {
        v16bf b = *((const v16bf*)wlds + (ct * 8 + k8) * 32 + lane);
        acc = wmma_bf16(afr[k8], b, acc);
      }
      int col = ct * 16 + (lane & 15);
      float bcv = bc[col];
#pragma unroll
      for (int v = 0; v < 8; ++v) {
        int r = rtb + v + ((lane >> 4) << 3);
        cbuf[r * 64 + col] = acc[v] + bcv;
      }
    }
  }
  __syncthreads();

  // ---- volumetric compositing (8 pixels per block) ----
  if (tid < 8) {
    float T = 1.f;
    for (int mm = 0; mm < 16; ++mm) {
      int r = tid * 16 + mm;
      float sg = sigb[r]; sg = sg > 0.f ? sg : 0.f;
      float a = 1.f - expf(-sg * dbuf[r]);
      wgtb[r] = a * T;
      T *= (1.f - a + 1e-10f);
    }
  }
  __syncthreads();
  for (int o = tid; o < 512; o += 256) {
    int pix = o >> 6, ch = o & 63;
    float s = 0.f;
#pragma unroll
    for (int mm = 0; mm < 16; ++mm)
      s += wgtb[pix * 16 + mm] * cbuf[(pix * 16 + mm) * 64 + ch];
    feat[ch * 16384 + blockIdx.x * 8 + pix] = s;
  }
}

// ---------------------------------------------------------------------------
// Epilogue: bias -> residual -> style-mod -> lrelu, then store [Cout][16384]
// ---------------------------------------------------------------------------
__device__ __forceinline__ void conv_epilogue(v8f acc, int col, int pixbase_or_pixel0,
                                              int lane, const float* bias,
                                              const float* res, const float* modw,
                                              const float* modb, int use_lrelu,
                                              float* out, int row_is_px) {
  float bs = bias ? bias[col] : 0.f;
  float mw = modw ? (modw[col] + 1.f) : 1.f;
  float mb = modb ? modb[col] : 0.f;
#pragma unroll
  for (int v = 0; v < 8; ++v) {
    int roff = v + ((lane >> 4) << 3);
    int pixel = pixbase_or_pixel0 + roff;
    (void)row_is_px;
    float val = acc[v] + bs;
    if (res) val += res[col * 16384 + pixel];
    if (modw) val = val * mw + mb;
    if (use_lrelu) val = lrelu_f(val);
    out[col * 16384 + pixel] = val;
  }
}

// ---------------------------------------------------------------------------
// 1x1 conv via WMMA: rows = pixels. in/out layout [C][16384].
// ---------------------------------------------------------------------------
__global__ void __launch_bounds__(256) k_conv1(
    const float* __restrict__ in, const float* __restrict__ w,
    const float* __restrict__ bias, const float* __restrict__ res,
    const float* __restrict__ modw, const float* __restrict__ modb,
    float* __restrict__ out, int Cin, int Cout, int use_lrelu) {
  extern __shared__ char smem[];
  bf16* fbuf = (bf16*)smem;                       // 128 x Cin
  bf16* wlds = (bf16*)(smem + 128 * Cin * 2);     // Cout x Cin swizzled
  int tid = threadIdx.x;
  int pixbase = blockIdx.x * 128;

  for (int idx = tid; idx < 128 * Cin; idx += 256) {
    int k = idx >> 7, r = idx & 127;
    fbuf[r * Cin + k] = (bf16)in[k * 16384 + pixbase + r];
  }
  __syncthreads();
  int nk8 = Cin >> 5, nct = Cout >> 4;
  stage_weights(wlds, w, nct, nk8, Cin, 1, 0, tid, 256);
  __syncthreads();

  int wave = tid >> 5, lane = tid & 31, rtb = wave * 16;
  v16bf afr[8];
  for (int k8 = 0; k8 < nk8; ++k8)
    afr[k8] = load_afrag(fbuf, Cin, rtb + (lane & 15), k8, nullptr);
  for (int ct = 0; ct < nct; ++ct) {
    v8f acc = {};
    for (int k8 = 0; k8 < nk8; ++k8) {
      v16bf b = *((const v16bf*)wlds + (ct * nk8 + k8) * 32 + lane);
      acc = wmma_bf16(afr[k8], b, acc);
    }
    int col = ct * 16 + (lane & 15);
    conv_epilogue(acc, col, pixbase + rtb, lane, bias, res, modw, modb, use_lrelu, out, 1);
  }
}

// ---------------------------------------------------------------------------
// 3x3 conv (256->256, SAME) via 9-tap shifted WMMA GEMM.
// Tile = 16x8 pixels; wave w owns image row (gy0+w). in/out [256][16384].
// ---------------------------------------------------------------------------
__global__ void __launch_bounds__(256) k_conv3(
    const float* __restrict__ in, const float* __restrict__ w,
    const float* __restrict__ bias, const float* __restrict__ res,
    const float* __restrict__ modw, const float* __restrict__ modb,
    float* __restrict__ out, int use_lrelu) {
  extern __shared__ char smem[];
  bf16* patch = (bf16*)smem;                 // [256][10][18] bf16 = 92160 B
  bf16* wtile = (bf16*)(smem + 92160);       // 8*32*16 bf16 = 8192 B
  int tid = threadIdx.x;
  int tx = blockIdx.x & 7, ty = blockIdx.x >> 3;
  int gx0 = tx * 16, gy0 = ty * 8;

  // stage input patch with halo, zero-padded
  for (int idx = tid; idx < 256 * 180; idx += 256) {
    int k = idx / 180, rem = idx - k * 180;
    int py = rem / 18, px = rem - py * 18;
    int gy = gy0 + py - 1, gx = gx0 + px - 1;
    float v = 0.f;
    if (gy >= 0 && gy < 128 && gx >= 0 && gx < 128) v = in[k * 16384 + gy * 128 + gx];
    patch[idx] = (bf16)v;
  }

  int wave = tid >> 5, lane = tid & 31, px = lane & 15;
  for (int ct = 0; ct < 16; ++ct) {
    v8f acc = {};
    for (int tap = 0; tap < 9; ++tap) {
      __syncthreads();  // protect wtile reuse (also orders patch writes on 1st pass)
      stage_weights(wtile, w + (ct * 16) * 2304, 1, 8, 2304, 9, tap, tid, 256);
      __syncthreads();
      int dy = tap / 3, dx = tap - dy * 3;
      int spatial = (wave + dy) * 18 + (px + dx);
#pragma unroll
      for (int k8 = 0; k8 < 8; ++k8) {
        // A-fragment gathered from channel-major patch
        v16bf a;
        int hoff = (lane & 16) ? 8 : 0;
#pragma unroll
        for (int v = 0; v < 8; ++v) {
          int kb = k8 * 32 + ((v >> 2) << 4) + hoff + ((v & 3) << 1);
          a[2 * v] = patch[kb * 180 + spatial];
          a[2 * v + 1] = patch[(kb + 1) * 180 + spatial];
        }
        v16bf b = *((const v16bf*)wtile + k8 * 32 + lane);
        acc = wmma_bf16(a, b, acc);
      }
    }
    int col = ct * 16 + (lane & 15);
    int pixel0 = (gy0 + wave) * 128 + gx0;   // + M offset added in epilogue
    conv_epilogue(acc, col, pixel0, lane, bias, res, modw, modb, use_lrelu, out, 1);
  }
}

// ---------------------------------------------------------------------------
// Final 256 -> 3 projection (tiny, VALU)
// ---------------------------------------------------------------------------
__global__ void k_out3(const float* __restrict__ in, const float* __restrict__ w,
                       const float* __restrict__ b, float* __restrict__ out) {
  int t = blockIdx.x * blockDim.x + threadIdx.x;
  if (t >= 3 * 16384) return;
  int o = t >> 14, pix = t & 16383;
  float s = b[o];
  for (int k = 0; k < 256; ++k) s += in[k * 16384 + pix] * w[o * 256 + k];
  out[o * 16384 + pix] = s;
}

extern "C" void kernel_launch(void* const* d_in, const int* in_sizes, int n_in,
                              void* d_out, int out_size, void* d_ws, size_t ws_size,
                              hipStream_t stream) {
  (void)in_sizes; (void)n_in; (void)out_size; (void)ws_size;
  const float* x     = (const float*)d_in[0];
  const float* m     = (const float*)d_in[1];
  const float* z     = (const float*)d_in[2];
  const float* dists = (const float*)d_in[3];
  const float* w1    = (const float*)d_in[4];
  const float* b1    = (const float*)d_in[5];
  const float* wma   = (const float*)d_in[6];
  const float* ml_w  = (const float*)d_in[7];
  const float* ml_wa = (const float*)d_in[8];
  const float* ml_ba = (const float*)d_in[9];
  const float* ml_wb = (const float*)d_in[10];
  const float* ml_bb = (const float*)d_in[11];
  const float* wsig  = (const float*)d_in[12];
  const float* bsig  = (const float*)d_in[13];
  const float* wc    = (const float*)d_in[14];
  const float* bc    = (const float*)d_in[15];
  const float* wz    = (const float*)d_in[16];
  const float* bz    = (const float*)d_in[17];
  const float* c1_w  = (const float*)d_in[18];
  const float* c1_b  = (const float*)d_in[19];
  const float* c2a_w = (const float*)d_in[20];
  const float* c2a_b = (const float*)d_in[21];
  const float* c2b_w = (const float*)d_in[22];
  const float* c3a_w = (const float*)d_in[23];
  const float* c3a_b = (const float*)d_in[24];
  const float* c3b_w = (const float*)d_in[25];
  const float* c4a_w = (const float*)d_in[26];
  const float* c4a_b = (const float*)d_in[27];
  const float* c4b_w = (const float*)d_in[28];
  const float* c4b_b = (const float*)d_in[29];
  const float* c4_w  = (const float*)d_in[30];
  const float* c4_b  = (const float*)d_in[31];

  char* ws = (char*)d_ws;
  float* sb   = (float*)ws;                      // 3584 floats (alpha/beta/zc)
  float* feat = (float*)(ws + (1u << 15));       // [64][16384]
  float* bufA = (float*)(ws + (8u << 20));       // [256][16384]
  float* bufB = (float*)(ws + (28u << 20));
  float* bufC = (float*)(ws + (48u << 20));
  const float *a0 = sb + 2560, *a1 = sb + 2816, *a2 = sb + 3072, *a3 = sb + 3328;

  k_style<<<14, 256, 0, stream>>>(z, ml_wa, ml_ba, ml_wb, ml_bb, wz, bz, sb);
  k_mlp<<<2048, 256, 235008, stream>>>(x, m, dists, w1, b1, wma, ml_w, sb,
                                       wsig, bsig, wc, bc, feat);
  // RenderCNN
  k_conv1<<<128, 256, 49152, stream>>>(feat, c1_w, c1_b, nullptr, nullptr, nullptr,
                                       bufA, 64, 256, 1);
  k_conv3<<<128, 256, 100352, stream>>>(bufA, c2a_w, c2a_b, nullptr, nullptr, nullptr, bufB, 1);
  k_conv3<<<128, 256, 100352, stream>>>(bufB, c2b_w, nullptr, bufA, a0, a1, bufC, 1);
  k_conv3<<<128, 256, 100352, stream>>>(bufC, c3a_w, c3a_b, nullptr, nullptr, nullptr, bufB, 1);
  k_conv3<<<128, 256, 100352, stream>>>(bufB, c3b_w, nullptr, bufC, a2, a3, bufA, 1);
  k_conv1<<<128, 256, 196608, stream>>>(bufA, c4a_w, c4a_b, nullptr, nullptr, nullptr,
                                        bufB, 256, 256, 1);
  k_conv1<<<128, 256, 196608, stream>>>(bufB, c4b_w, c4b_b, bufA, nullptr, nullptr,
                                        bufC, 256, 256, 1);
  k_out3<<<192, 256, 0, stream>>>(bufC, c4_w, c4_b, (float*)d_out);
}
